// LPFormer_54889682043696
// MI455X (gfx1250) — compile-verified, matrix operand order checked by reading, not run on hardware
//
#include <hip/hip_runtime.h>
#include <hip/hip_bf16.h>

// ---------------- model constants ----------------
constexpr int Bn   = 4;
constexpr int Np   = 2048;
constexpr int NKp  = 14;
constexpr int EMB  = 256;
constexpr int NHh  = 8;
constexpr int HDim = 32;
constexpr int VSv  = 64;
constexpr int Sreal = NKp + Np;     // 2062
constexpr int Sp    = 2080;         // padded to multiple of 32
constexpr int Mtok  = Bn * Sp;      // 8320
constexpr int Mpts  = Bn * Np;      // 8192

// ---------------- WMMA plumbing (CDNA5, wave32) ----------------
typedef __attribute__((ext_vector_type(8)))  __bf16 v8bf;
typedef __attribute__((ext_vector_type(16))) __bf16 v16bf;
typedef __attribute__((ext_vector_type(8)))  float  v8f;
typedef __attribute__((ext_vector_type(4)))  unsigned v4u;
typedef __attribute__((ext_vector_type(8)))  int    v8i;
typedef __attribute__((ext_vector_type(4)))  int    v4i;

// A/B fragment for v_wmma_f32_16x16x32_bf16 (ISA 7.12.2):
// lane holds row (A) / col (B) = lane&15; K chunks: [hi*8 .. hi*8+7] and
// [16+hi*8 .. 16+hi*8+7] where hi = lane>>4.  Caller pre-adds hi*8; we load
// 8 bf16 at p and 8 bf16 at p+16 (two global_load_b128 / ds_load_b128).
__device__ inline v16bf load_frag(const __bf16* p) {
  v8bf lo = *(const v8bf*)(p);
  v8bf hh = *(const v8bf*)(p + 16);
  v16bf r;
#pragma unroll
  for (int i = 0; i < 8; ++i) { r[i] = lo[i]; r[i + 8] = hh[i]; }
  return r;
}

__device__ inline v8f wmma_bf16(v16bf a, v16bf b, v8f c) {
  return __builtin_amdgcn_wmma_f32_16x16x32_bf16(false, a, false, b, (short)0, c,
                                                 false, false);
}

__device__ inline float clampf(float x, float a, float b) {
  return fminf(fmaxf(x, a), b);
}

// ---------------- Tensor Data Mover (gfx1250) ----------------
#if defined(__has_builtin)
#  if __has_builtin(__builtin_amdgcn_tensor_load_to_lds)
#    define LP_HAVE_TDM 1
#  endif
#endif
#if __has_include(<hip/amd_detail/amd_gfx1250_TDM.h>)
#  define LP_TDM_6ARG 1
#endif

#if defined(LP_HAVE_TDM)
// One-descriptor 1-D DMA: copy `ndw` dwords from global to LDS (ISA ch.8 D#).
// group0: count=1 | lds_addr | global_addr(57b) | type=2 ("image")
// group1: data_size=4B, tensor_dim0=tile_dim0=ndw, tile_dim1=tensor_dim1=1,
//         tensor_dim0_stride=ndw; no multicast/pad/iterate/gather.
__device__ inline void tdm_load_1d(const void* gsrc, void* lds_dst,
                                   unsigned ndw) {
  unsigned lds_off = (unsigned)(uintptr_t)lds_dst;   // flat LDS addr[31:0]
  unsigned long long ga = (unsigned long long)(uintptr_t)gsrc;
  v4u g0;
  g0[0] = 1u;                                        // count=1, user mode
  g0[1] = lds_off;                                   // lds_addr
  g0[2] = (unsigned)ga;                              // global_addr[31:0]
  g0[3] = (unsigned)((ga >> 32) & 0x1FFFFFFu) | (2u << 30);  // [56:32] | type=2
  v8i g1;
  g1[0] = (int)(2u << 16);                           // data_size = 4 bytes
  g1[1] = (int)((ndw & 0xFFFFu) << 16);              // tensor_dim0[15:0]
  g1[2] = (int)(((ndw >> 16) & 0xFFFFu) | (1u << 16));  // td0[31:16] | td1=1
  g1[3] = (int)((ndw & 0xFFFFu) << 16);              // tile_dim0 (<=65535)
  g1[4] = 1;                                         // tile_dim1=1, tile_dim2=0
  g1[5] = (int)ndw;                                  // td0_stride[31:0]
  g1[6] = 0;                                         // td0_stride[47:32], td1_s
  g1[7] = 0;
  v4i z4 = {0, 0, 0, 0};
#if defined(LP_TDM_6ARG)
  v8i z8 = {0, 0, 0, 0, 0, 0, 0, 0};
  __builtin_amdgcn_tensor_load_to_lds(g0, g1, z4, z4, z8, 0);
#else
  __builtin_amdgcn_tensor_load_to_lds(g0, g1, z4, z4, 0);
#endif
}
#endif  // LP_HAVE_TDM

// ---------------- generic tiled WMMA GEMM ----------------
// C[M,N] = A[M,K](bf16) * Bt[N,K](bf16)^T + bias[N]; optional relu; writes
// f32 and/or bf16 outputs. K%32==0, N%(16*NT)==0.  256 threads = 8 waves;
// wave w -> row tile blockIdx.y*8+w; each wave computes 16 x (16*NT) with the
// A fragment reused across NT WMMAs (NT independent accumulator chains keeps
// the XDL pipe free of D->C RAW stalls, ISA 7.12.1).
template <int NT>
__global__ __launch_bounds__(256) void wmma_gemm(
    const __bf16* __restrict__ A, const __bf16* __restrict__ Bt,
    const float* __restrict__ bias, float* Cf, __bf16* Ch,
    int M, int N, int K, int relu) {
  int wave = threadIdx.x >> 5;
  int lane = threadIdx.x & 31;
  int tm = blockIdx.y * 8 + wave;
  if (tm * 16 >= M) return;                 // wave-uniform
  int ln = lane & 15;
  int hi = lane >> 4;
  int tn0 = blockIdx.x * NT;

  const __bf16* ap = A + (size_t)(tm * 16 + ln) * K + hi * 8;
  const __bf16* bp[NT];
  v8f c[NT];
#pragma unroll
  for (int t = 0; t < NT; ++t) {
    bp[t] = Bt + (size_t)((tn0 + t) * 16 + ln) * K + hi * 8;
#pragma unroll
    for (int j = 0; j < 8; ++j) c[t][j] = 0.f;
  }

  for (int k0 = 0; k0 < K; k0 += 32) {
    v16bf a = load_frag(ap + k0);
    if (k0 + 32 < K) __builtin_prefetch(ap + k0 + 32, 0, 1);  // global_prefetch
#pragma unroll
    for (int t = 0; t < NT; ++t) {
      v16bf b = load_frag(bp[t] + k0);
      if (k0 + 32 < K) __builtin_prefetch(bp[t] + k0 + 32, 0, 1);
      c[t] = wmma_bf16(a, b, c[t]);
    }
  }

#pragma unroll
  for (int t = 0; t < NT; ++t) {
    int ncol = (tn0 + t) * 16 + ln;
    float bv = bias ? bias[ncol] : 0.f;
#pragma unroll
    for (int v = 0; v < 8; ++v) {
      int r = tm * 16 + hi * 8 + v;         // D layout: vgpr v -> row hi*8+v
      float val = c[t][v] + bv;
      if (relu) val = fmaxf(val, 0.f);
      size_t oi = (size_t)r * N + ncol;
      if (Cf) Cf[oi] = val;
      if (Ch) Ch[oi] = (__bf16)val;
    }
  }
}

// ---------------- weight prep: fp32 (fi,fo) -> bf16 transposed (fo,Kp) ------
__global__ void prep_wt(const float* __restrict__ W, __bf16* __restrict__ Wt,
                        int fi, int fo, int Kp) {
  int t = blockIdx.x * blockDim.x + threadIdx.x;
  if (t >= fo * Kp) return;
  int o = t / Kp, k = t % Kp;
  Wt[t] = (k < fi) ? (__bf16)W[(size_t)k * fo + o] : (__bf16)0.f;
}

// ---------------- misc elementwise kernels ----------------
__global__ void fill_f32(float* p, size_t n) {
  size_t t = (size_t)blockIdx.x * blockDim.x + threadIdx.x;
  if (t < n) p[t] = 0.f;
}

__global__ void voxelize(const float* __restrict__ pts,
                         const float* __restrict__ ctr, float* vox) {
  int t = blockIdx.x * blockDim.x + threadIdx.x;
  if (t >= Mpts) return;
  int b = t / Np;
  int idx[3];
#pragma unroll
  for (int c = 0; c < 3; ++c) {
    float loc = (pts[(size_t)t * 3 + c] - ctr[b * 3 + c] + 1.f) * 0.5f;
    idx[c] = (int)clampf(loc * (float)VSv, 0.f, (float)(VSv - 1));
  }
  vox[(((size_t)b * VSv + idx[0]) * VSv + idx[1]) * VSv + idx[2]] = 1.0f;
}

// ---------------- fused conv1->conv2->grid_sample + feature assembly --------
// One thread per point: evaluate relu(conv2(relu(conv1(vox)))) only at the
// <=8 trilinear corner voxels, then write the 64-wide padded MLP input row
// [pts(3), local(3), voxfeat(32), 0...0] in bf16.  conv2 weights (55KB) are
// staged into LDS by the Tensor Data Mover when available.
__global__ __launch_bounds__(256) void conv_sample(
    const float* __restrict__ pts, const float* __restrict__ ctr,
    const float* __restrict__ vox, const float* __restrict__ w1,
    const float* __restrict__ b1, const float* __restrict__ w2,
    const float* __restrict__ b2, __bf16* __restrict__ A0) {
  __shared__ float sw1[16 * 27];
  __shared__ float sw2[32 * 16 * 27];
  __shared__ float sb1[16];
  __shared__ float sb2[32];
#if defined(LP_HAVE_TDM)
  if (threadIdx.x < 32) {                   // one wave issues the DMA
    tdm_load_1d(w2, sw2, 32u * 16u * 27u);
    __builtin_amdgcn_s_wait_tensorcnt(0);
  }
#else
  for (int i = threadIdx.x; i < 32 * 16 * 27; i += 256) sw2[i] = w2[i];
#endif
  for (int i = threadIdx.x; i < 16 * 27; i += 256) sw1[i] = w1[i];
  if (threadIdx.x < 16) sb1[threadIdx.x] = b1[threadIdx.x];
  if (threadIdx.x < 32) sb2[threadIdx.x] = b2[threadIdx.x];
  __syncthreads();

  int t = blockIdx.x * 256 + threadIdx.x;
  if (t >= Mpts) return;
  int b = t / Np;

  float p[3], lcl[3], loc[3];
#pragma unroll
  for (int c = 0; c < 3; ++c) {
    p[c]   = pts[(size_t)t * 3 + c];
    lcl[c] = p[c] - ctr[b * 3 + c];
    loc[c] = (lcl[c] + 1.f) * 0.5f;
  }
  // grid reversed: gx<-loc[2] (W), gy<-loc[1] (H), gz<-loc[0] (D)
  float gx = clampf(2.f * loc[2] - 1.f, -1.1f, 1.1f);
  float gy = clampf(2.f * loc[1] - 1.f, -1.1f, 1.1f);
  float gz = clampf(2.f * loc[0] - 1.f, -1.1f, 1.1f);
  float ix = ((gx + 1.f) * 64.f - 1.f) * 0.5f;
  float iy = ((gy + 1.f) * 64.f - 1.f) * 0.5f;
  float iz = ((gz + 1.f) * 64.f - 1.f) * 0.5f;
  float x0 = floorf(ix), y0 = floorf(iy), z0 = floorf(iz);
  float wx1 = ix - x0, wy1 = iy - y0, wz1 = iz - z0;

  const float* vb = vox + (size_t)b * VSv * VSv * VSv;
  float outf[32];
#pragma unroll
  for (int oc = 0; oc < 32; ++oc) outf[oc] = 0.f;

  for (int dz = 0; dz < 2; ++dz)
    for (int dy = 0; dy < 2; ++dy)
      for (int dx = 0; dx < 2; ++dx) {
        int xi = (int)x0 + dx, yi = (int)y0 + dy, zi = (int)z0 + dz;
        float wv = (dx ? wx1 : 1.f - wx1) * (dy ? wy1 : 1.f - wy1) *
                   (dz ? wz1 : 1.f - wz1);
        bool valid = (xi >= 0) && (xi < 64) && (yi >= 0) && (yi < 64) &&
                     (zi >= 0) && (zi < 64);
        if (!valid) continue;               // masked contribution is 0

        float acc2[32];
#pragma unroll
        for (int oc = 0; oc < 32; ++oc) acc2[oc] = 0.f;

        for (int t0 = 0; t0 < 3; ++t0)
          for (int t1 = 0; t1 < 3; ++t1)
            for (int t2 = 0; t2 < 3; ++t2) {
              int n0 = zi + t0 - 1, n1 = yi + t1 - 1, n2 = xi + t2 - 1;
              float v27[27];
              int u = 0;
#pragma unroll
              for (int u0 = 0; u0 < 3; ++u0)
#pragma unroll
                for (int u1 = 0; u1 < 3; ++u1)
#pragma unroll
                  for (int u2 = 0; u2 < 3; ++u2) {
                    int q0 = n0 + u0 - 1, q1 = n1 + u1 - 1, q2 = n2 + u2 - 1;
                    bool ok = (q0 >= 0) && (q0 < 64) && (q1 >= 0) && (q1 < 64) &&
                              (q2 >= 0) && (q2 < 64);
                    v27[u++] = ok ? vb[((size_t)q0 * 64 + q1) * 64 + q2] : 0.f;
                  }
              int tap = (t0 * 3 + t1) * 3 + t2;
              for (int ic = 0; ic < 16; ++ic) {
                float s = sb1[ic];
#pragma unroll
                for (int uu = 0; uu < 27; ++uu) s += sw1[ic * 27 + uu] * v27[uu];
                float hval = fmaxf(s, 0.f);
                if (hval != 0.f) {
#pragma unroll
                  for (int oc = 0; oc < 32; ++oc)
                    acc2[oc] += sw2[(oc * 16 + ic) * 27 + tap] * hval;
                }
              }
            }
#pragma unroll
        for (int oc = 0; oc < 32; ++oc)
          outf[oc] += fmaxf(acc2[oc] + sb2[oc], 0.f) * wv;
      }

  __bf16* ar = A0 + (size_t)t * 64;
#pragma unroll
  for (int c = 0; c < 3; ++c) { ar[c] = (__bf16)p[c]; ar[3 + c] = (__bf16)lcl[c]; }
#pragma unroll
  for (int j = 0; j < 32; ++j) ar[6 + j] = (__bf16)outf[j];
#pragma unroll
  for (int j = 38; j < 64; ++j) ar[j] = (__bf16)0.f;
}

// ---------------- token assembly: X0 = [kpt_emb ; mlp(points) ; 0-pad] ------
__global__ void assemble_x(const float* __restrict__ kpt,
                           const float* __restrict__ f, float* __restrict__ X,
                           __bf16* __restrict__ Xb) {
  size_t t = (size_t)blockIdx.x * blockDim.x + threadIdx.x;
  if (t >= (size_t)Mtok * EMB) return;
  int c = (int)(t % EMB);
  int r = (int)(t / EMB);
  int b = r / Sp, s = r % Sp;
  float v = 0.f;
  if (s < NKp)        v = kpt[(size_t)s * EMB + c];
  else if (s < Sreal) v = f[((size_t)(b * Np + (s - NKp))) * EMB + c];
  X[t]  = v;
  Xb[t] = (__bf16)v;
}

// ---------------- V transpose: (b,s,h,d) -> Vt[(b,h,d), s] -----------------
__global__ void transpose_v(const __bf16* __restrict__ V,
                            __bf16* __restrict__ Vt) {
  size_t t = (size_t)blockIdx.x * blockDim.x + threadIdx.x;
  if (t >= (size_t)Mtok * EMB) return;
  int c = (int)(t % EMB);
  int r = (int)(t / EMB);
  int b = r / Sp, s = r % Sp;
  int h = c / HDim, d = c % HDim;
  Vt[((size_t)((b * NHh + h) * HDim + d)) * Sp + s] = V[t];
}

// ---------------- wave-level flash attention (one wave = 16 queries) --------
// scores: Q(16x32hd) x K^T(32hd x 32keys) = 2 WMMAs; online softmax across the
// 16-lane C-layout groups; P transposed via 1KB LDS; ctx: P(16x32) x V(32x32)
// = 2 WMMAs against pre-transposed Vt. Residual update of X/Xb is in place
// (each (b,h,qtile) owns a disjoint 16x32 slice).
__global__ __launch_bounds__(32) void flash_attn(
    const __bf16* __restrict__ Qb, const __bf16* __restrict__ Kb,
    const __bf16* __restrict__ Vt, float* __restrict__ X,
    __bf16* __restrict__ Xb) {
  __shared__ __bf16 pls[16][32];
  int lane = threadIdx.x & 31;
  int qt = blockIdx.x, h = blockIdx.y, b = blockIdx.z;
  int ln = lane & 15, hi = lane >> 4;
  const float scl = 0.17677669529663687f;   // 1/sqrt(32)

  v16bf afrag = load_frag(Qb + ((size_t)(b * Sp + qt * 16 + ln)) * EMB +
                          h * HDim + hi * 8);
  float m[8], l[8];
  v8f acc0 = {}, acc1 = {};
#pragma unroll
  for (int v = 0; v < 8; ++v) { m[v] = -1e30f; l[v] = 0.f; }

  for (int kb = 0; kb < Sp; kb += 32) {
    int kk0 = kb + ln, kk1 = kb + 16 + ln;
    v16bf bk0 = load_frag(Kb + (size_t)(b * Sp + kk0) * EMB + h * HDim + hi * 8);
    v16bf bk1 = load_frag(Kb + (size_t)(b * Sp + kk1) * EMB + h * HDim + hi * 8);
    v8f s0 = {}, s1 = {};
    s0 = wmma_bf16(afrag, bk0, s0);
    s1 = wmma_bf16(afrag, bk1, s1);
    bool ok0 = kk0 < Sreal, ok1 = kk1 < Sreal;

#pragma unroll
    for (int v = 0; v < 8; ++v) {
      float e0 = ok0 ? s0[v] * scl : -1e30f;
      float e1 = ok1 ? s1[v] * scl : -1e30f;
      float mx = fmaxf(e0, e1);
#pragma unroll
      for (int msk = 8; msk >= 1; msk >>= 1)
        mx = fmaxf(mx, __shfl_xor(mx, msk, 32));
      float mn = fmaxf(m[v], mx);
      float p0 = __expf(e0 - mn);
      float p1 = __expf(e1 - mn);
      float rs = p0 + p1;
#pragma unroll
      for (int msk = 8; msk >= 1; msk >>= 1) rs += __shfl_xor(rs, msk, 32);
      float scale = __expf(m[v] - mn);
      m[v] = mn;
      l[v] = l[v] * scale + rs;
      acc0[v] *= scale;
      acc1[v] *= scale;
      pls[hi * 8 + v][ln]      = (__bf16)p0;   // D-layout -> row-major P tile
      pls[hi * 8 + v][16 + ln] = (__bf16)p1;
    }
    __syncthreads();  // single wave: lowers to waits (+S_NOP barrier)

    v16bf pa  = load_frag(&pls[ln][hi * 8]);
    v16bf vb0 = load_frag(Vt + ((size_t)((b * NHh + h) * HDim + ln)) * Sp +
                          kb + hi * 8);
    v16bf vb1 = load_frag(Vt + ((size_t)((b * NHh + h) * HDim + 16 + ln)) * Sp +
                          kb + hi * 8);
    acc0 = wmma_bf16(pa, vb0, acc0);
    acc1 = wmma_bf16(pa, vb1, acc1);
    __syncthreads();
  }

#pragma unroll
  for (int v = 0; v < 8; ++v) {
    int q = qt * 16 + hi * 8 + v;
    float inv = 1.f / l[v];
    size_t base = (size_t)(b * Sp + q) * EMB + h * HDim;
    float o0 = X[base + ln]      + acc0[v] * inv;
    float o1 = X[base + 16 + ln] + acc1[v] * inv;
    X[base + ln] = o0;           X[base + 16 + ln] = o1;
    Xb[base + ln] = (__bf16)o0;  Xb[base + 16 + ln] = (__bf16)o1;
  }
}

// ---------------- keypoint heads (xy / z / vis) ----------------
__device__ inline void head3(const float* xr, const float* w0, const float* c0,
                             const float* w1, const float* c1, const float* w2,
                             const float* c2, int no, float* o) {
  float h1[16];
  for (int j = 0; j < 16; ++j) {
    float s = c0[j];
    for (int c = 0; c < 256; ++c) s += xr[c] * w0[(size_t)c * 16 + j];
    h1[j] = fmaxf(s, 0.f);
  }
  float h2[8];
  for (int j = 0; j < 8; ++j) {
    float s = c1[j];
    for (int c = 0; c < 16; ++c) s += h1[c] * w1[c * 8 + j];
    h2[j] = s;
  }
  for (int j = 0; j < no; ++j) {
    float s = c2[j];
    for (int c = 0; c < 8; ++c) s += h2[c] * w2[c * no + j];
    o[j] = s;
  }
}

__global__ void kp_heads(const float* __restrict__ X, const float* xw0,
                         const float* xb0, const float* xw1, const float* xb1,
                         const float* xw2, const float* xb2, const float* zw0,
                         const float* zb0, const float* zw1, const float* zb1,
                         const float* zw2, const float* zb2, const float* vw0,
                         const float* vb0, const float* vw1, const float* vb1,
                         const float* vw2, const float* vb2,
                         float* __restrict__ out) {
  int t = threadIdx.x;
  if (t >= Bn * NKp) return;
  int b = t / NKp, k = t % NKp;
  const float* xr = X + (size_t)(b * Sp + k) * EMB;
  float xy[2], zz[1], vv[2];
  head3(xr, xw0, xb0, xw1, xb1, xw2, xb2, 2, xy);
  head3(xr, zw0, zb0, zw1, zb1, zw2, zb2, 1, zz);
  head3(xr, vw0, vb0, vw1, vb1, vw2, vb2, 2, vv);
  out[(size_t)t * 3 + 0] = xy[0];
  out[(size_t)t * 3 + 1] = xy[1];
  out[(size_t)t * 3 + 2] = zz[0];
  float mx = fmaxf(vv[0], vv[1]);
  float e0 = __expf(vv[0] - mx), e1 = __expf(vv[1] - mx);
  out[168 + t] = e0 / (e0 + e1);
}

// ---------------- seg final layer (16 -> 15) + softmax ----------------
__global__ void seg_final(const __bf16* __restrict__ H2,
                          const float* __restrict__ w2,
                          const float* __restrict__ b2,
                          float* __restrict__ out) {
  int t = blockIdx.x * blockDim.x + threadIdx.x;
  if (t >= Mpts) return;
  int b = t / Np, i = t % Np;
  const __bf16* hr = H2 + (size_t)(b * Sp + NKp + i) * 16;
  float h[16];
#pragma unroll
  for (int j = 0; j < 16; ++j) h[j] = (float)hr[j];
  float lg[15], mx = -1e30f;
#pragma unroll
  for (int c = 0; c < 15; ++c) {
    float s = b2[c];
#pragma unroll
    for (int j = 0; j < 16; ++j) s += h[j] * w2[j * 15 + c];
    lg[c] = s;
    mx = fmaxf(mx, s);
  }
  float den = 0.f;
#pragma unroll
  for (int c = 0; c < 15; ++c) { lg[c] = __expf(lg[c] - mx); den += lg[c]; }
  float inv = 1.f / den;
#pragma unroll
  for (int c = 0; c < 15; ++c) out[224 + (size_t)t * 15 + c] = lg[c] * inv;
}

// ---------------- host side ----------------
extern "C" void kernel_launch(void* const* d_in, const int* in_sizes, int n_in,
                              void* d_out, int out_size, void* d_ws,
                              size_t ws_size, hipStream_t stream) {
  (void)in_sizes; (void)n_in; (void)out_size; (void)ws_size;
  const float* pts = (const float*)d_in[0];
  const float* ctr = (const float*)d_in[1];

  // params flattened in jax pytree (sorted-key) order
  int ix = 2;
  const float *attn_w[4][3], *attn_b[4][3];        // per layer: [k, q, v]
  for (int L = 0; L < 4; ++L)
    for (int j = 0; j < 3; ++j) {
      attn_w[L][j] = (const float*)d_in[ix++];
      attn_b[L][j] = (const float*)d_in[ix++];
    }
  const float* c1b = (const float*)d_in[ix++];
  const float* c1w = (const float*)d_in[ix++];
  const float* c2b = (const float*)d_in[ix++];
  const float* c2w = (const float*)d_in[ix++];
  const float* kpt = (const float*)d_in[ix++];
  const float *mw[3], *mb[3];
  for (int j = 0; j < 3; ++j) { mw[j] = (const float*)d_in[ix++]; mb[j] = (const float*)d_in[ix++]; }
  const float *sw[3], *sb[3];
  for (int j = 0; j < 3; ++j) { sw[j] = (const float*)d_in[ix++]; sb[j] = (const float*)d_in[ix++]; }
  const float *vw[3], *vb[3];
  for (int j = 0; j < 3; ++j) { vw[j] = (const float*)d_in[ix++]; vb[j] = (const float*)d_in[ix++]; }
  const float *xw[3], *xb[3];
  for (int j = 0; j < 3; ++j) { xw[j] = (const float*)d_in[ix++]; xb[j] = (const float*)d_in[ix++]; }
  const float *zw[3], *zb[3];
  for (int j = 0; j < 3; ++j) { zw[j] = (const float*)d_in[ix++]; zb[j] = (const float*)d_in[ix++]; }

  float* out = (float*)d_out;

  // ---- workspace layout (all 256B aligned; total ~76MB, L2-resident) ----
  char* w = (char*)d_ws;
  size_t off = 0;
  auto take = [&](size_t bytes) {
    char* p = w + off;
    off += (bytes + 255) & ~(size_t)255;
    return p;
  };
  float*  vox   = (float*)take((size_t)Bn * VSv * VSv * VSv * 4);
  __bf16* A0    = (__bf16*)take((size_t)Mpts * 64 * 2);
  __bf16* H1    = (__bf16*)take((size_t)Mpts * 1024 * 2);
  __bf16* H2    = (__bf16*)take((size_t)Mpts * 512 * 2);
  float*  Ff    = (float*)take((size_t)Mpts * EMB * 4);
  float*  X     = (float*)take((size_t)Mtok * EMB * 4);
  __bf16* Xb    = (__bf16*)take((size_t)Mtok * EMB * 2);
  __bf16* Qb    = (__bf16*)take((size_t)Mtok * EMB * 2);
  __bf16* Kb    = (__bf16*)take((size_t)Mtok * EMB * 2);
  __bf16* Vb    = (__bf16*)take((size_t)Mtok * EMB * 2);
  __bf16* Vt    = (__bf16*)take((size_t)Mtok * EMB * 2);
  __bf16* SegH1 = (__bf16*)take((size_t)Mtok * EMB * 2);
  __bf16* SegH2 = (__bf16*)take((size_t)Mtok * 16 * 2);
  __bf16* Wt_m0 = (__bf16*)take((size_t)1024 * 64 * 2);
  __bf16* Wt_m1 = (__bf16*)take((size_t)512 * 1024 * 2);
  __bf16* Wt_m2 = (__bf16*)take((size_t)256 * 512 * 2);
  __bf16* Wt_at[4][3];
  for (int L = 0; L < 4; ++L)
    for (int j = 0; j < 3; ++j)
      Wt_at[L][j] = (__bf16*)take((size_t)256 * 256 * 2);
  __bf16* Wt_s0 = (__bf16*)take((size_t)256 * 256 * 2);
  __bf16* Wt_s1 = (__bf16*)take((size_t)16 * 256 * 2);

  auto gemm = [&](const __bf16* A, const __bf16* Bt, const float* bias,
                  float* Cf, __bf16* Ch, int M, int N, int K, int relu) {
    if (N % 64 == 0) {
      dim3 grid(N / 64, (M + 127) / 128);
      wmma_gemm<4><<<grid, 256, 0, stream>>>(A, Bt, bias, Cf, Ch, M, N, K, relu);
    } else {
      dim3 grid(N / 16, (M + 127) / 128);
      wmma_gemm<1><<<grid, 256, 0, stream>>>(A, Bt, bias, Cf, Ch, M, N, K, relu);
    }
  };
  auto prep = [&](const float* W, __bf16* Wt, int fi, int fo, int Kp) {
    int n = fo * Kp;
    prep_wt<<<(n + 255) / 256, 256, 0, stream>>>(W, Wt, fi, fo, Kp);
  };

  // 0) weight prep (bf16 + transpose)
  prep(mw[0], Wt_m0, 38, 1024, 64);
  prep(mw[1], Wt_m1, 1024, 512, 1024);
  prep(mw[2], Wt_m2, 512, 256, 512);
  for (int L = 0; L < 4; ++L)
    for (int j = 0; j < 3; ++j) prep(attn_w[L][j], Wt_at[L][j], 256, 256, 256);
  prep(sw[0], Wt_s0, 256, 256, 256);
  prep(sw[1], Wt_s1, 256, 16, 256);

  // 1) voxel grid
  {
    size_t n = (size_t)Bn * VSv * VSv * VSv;
    fill_f32<<<(unsigned)((n + 255) / 256), 256, 0, stream>>>(vox, n);
    voxelize<<<(Mpts + 255) / 256, 256, 0, stream>>>(pts, ctr, vox);
  }

  // 2) fused conv1/conv2/grid-sample -> padded MLP input rows
  conv_sample<<<Mpts / 256, 256, 0, stream>>>(pts, ctr, vox, c1w, c1b, c2w, c2b,
                                              A0);

  // 3) MLP (pure linear chain): 38(->64) -> 1024 -> 512 -> 256
  gemm(A0, Wt_m0, mb[0], nullptr, H1, Mpts, 1024, 64, 0);
  gemm(H1, Wt_m1, mb[1], nullptr, H2, Mpts, 512, 1024, 0);
  gemm(H2, Wt_m2, mb[2], Ff, nullptr, Mpts, 256, 512, 0);

  // 4) token assembly
  {
    size_t n = (size_t)Mtok * EMB;
    assemble_x<<<(unsigned)((n + 255) / 256), 256, 0, stream>>>(kpt, Ff, X, Xb);
  }

  // 5) 4 attention layers
  for (int L = 0; L < 4; ++L) {
    gemm(Xb, Wt_at[L][1], attn_b[L][1], nullptr, Qb, Mtok, 256, 256, 0);  // q
    gemm(Xb, Wt_at[L][0], attn_b[L][0], nullptr, Kb, Mtok, 256, 256, 0);  // k
    gemm(Xb, Wt_at[L][2], attn_b[L][2], nullptr, Vb, Mtok, 256, 256, 0);  // v
    {
      size_t n = (size_t)Mtok * EMB;
      transpose_v<<<(unsigned)((n + 255) / 256), 256, 0, stream>>>(Vb, Vt);
    }
    dim3 fg(Sp / 16, NHh, Bn);
    flash_attn<<<fg, 32, 0, stream>>>(Qb, Kb, Vt, X, Xb);
  }

  // 6) seg branch: 256 -relu-> 256 -> 16 -> (VALU) 15 + softmax
  gemm(Xb, Wt_s0, sb[0], nullptr, SegH1, Mtok, 256, 256, 1);
  gemm(SegH1, Wt_s1, sb[1], nullptr, SegH2, Mtok, 16, 256, 0);
  seg_final<<<(Mpts + 255) / 256, 256, 0, stream>>>(SegH2, sw[2], sb[2], out);

  // 7) keypoint heads (xy, z, vis)
  kp_heads<<<1, 64, 0, stream>>>(X, xw[0], xb[0], xw[1], xb[1], xw[2], xb[2],
                                 zw[0], zb[0], zw[1], zb[1], zw[2], zb[2],
                                 vw[0], vb[0], vw[1], vb[1], vw[2], vb[2], out);
}
